// Engram_63067299774780
// MI455X (gfx1250) — compile-verified, matrix operand order checked by reading, not run on hardware
//
#include <hip/hip_runtime.h>

// ---------------------------------------------------------------------------
// MI455X (gfx1250) Engram kernel: gather -> dual GEMM (WMMA bf16) -> gating ->
// dilated depthwise conv.
// Round-4 change: __builtin_amdgcn_sched_barrier(0) between the B-fragment
// load burst and the WMMA burst, so the scheduler cannot sink loads to their
// consumers. Forces 16 in-flight global_load_b128 per K-step with partial
// s_wait_loadcnt instead of 8 serialized full waits.
// ---------------------------------------------------------------------------

typedef __attribute__((ext_vector_type(16))) __bf16 v16bf;
typedef __attribute__((ext_vector_type(8)))  __bf16 v8bf;
typedef __attribute__((ext_vector_type(8)))  float  v8f;

#define D_      1024
#define GD      2048
#define NTOT    3072            // 2048 key cols + 1024 value cols
#define TOK     16              // tokens (M rows) per GEMM block
#define NTOK    16384           // B*S
#define KTILES  32              // 1024 / 32
#define NTILES  192             // 3072 / 16
#define WAVES   24
#define THREADS (WAVES * 32)    // 768
#define TILES_PER_WAVE 8        // 192 / 24

// cumsum of vocab sizes (compile-time)
__constant__ long long c_offs[16] = {
    0, 49999, 100020, 150043, 200076, 250123, 300174, 350227,
    400296, 450373, 500460, 550553, 600654, 650765, 700884, 751007};

// ---------------------------------------------------------------------------
// K0: pack [key_kernel | value_kernel] (f32, row-major K x N) into bf16 tiles
// laid out in WMMA B-fragment order: tile (kt,nt) is 512 bf16; lane l owns 16
// consecutive elements; element e of lane l maps to
//   n = nt*16 + (l & 15)
//   k = kt*32 + (e<8 ? (l>>4)*8 + e : 16 + (l>>4)*8 + (e-8))
// mirroring the documented 16-bit A-matrix striping (CDNA5 ISA).
// ---------------------------------------------------------------------------
__global__ void k0_pack(const float* __restrict__ kk, const float* __restrict__ vk,
                        __bf16* __restrict__ Wb)
{
    int t = blockIdx.x * blockDim.x + threadIdx.x;
    if (t >= D_ * NTOT) return;
    int tile = t >> 9;              // /512
    int rem  = t & 511;
    int l    = rem >> 4;
    int e    = rem & 15;
    int kt   = tile / NTILES;
    int nt   = tile - kt * NTILES;
    int hlf  = l >> 4;
    int n    = nt * 16 + (l & 15);
    int k    = kt * 32 + ((e < 8) ? (hlf * 8 + e) : (16 + hlf * 8 + (e - 8)));
    float w  = (n < GD) ? kk[(size_t)k * GD + n]
                        : vk[(size_t)k * D_ + (n - GD)];
    Wb[t] = (__bf16)w;
}

// ---------------------------------------------------------------------------
// K1: multi-head hashed embedding gather -> bf16 A matrix [NTOK, 1024]
// ---------------------------------------------------------------------------
__global__ void k1_gather(const int* __restrict__ ids,
                          const float* __restrict__ table,
                          __bf16* __restrict__ Eb)
{
    int token = blockIdx.x;                 // 0..16383
    int e     = threadIdx.x * 4;            // 0..1020, 4 elems per thread
    int h     = e >> 6;                     // head 0..15
    int d     = e & 63;
    long long id = (long long)ids[token * 16 + h] + c_offs[h];
    const float4 v = *(const float4*)(table + id * 64 + d);
    __bf16* dst = Eb + (size_t)token * D_ + e;
    dst[0] = (__bf16)v.x;
    dst[1] = (__bf16)v.y;
    dst[2] = (__bf16)v.z;
    dst[3] = (__bf16)v.w;
}

// ---------------------------------------------------------------------------
// K2: per 16-token stripe: C[16,3072] = A[16,1024] x W[1024,3072] via
// v_wmma_f32_16x16x32_bf16, C parked in LDS, fused gating epilogue.
// ---------------------------------------------------------------------------
__global__ __launch_bounds__(THREADS) void
k2_gemm(const __bf16* __restrict__ Eb, const __bf16* __restrict__ Wb,
        const float* __restrict__ hidden,
        const float* __restrict__ key_bias, const float* __restrict__ value_bias,
        const float* __restrict__ k_scale,  const float* __restrict__ q_scale,
        float* __restrict__ out, __bf16* __restrict__ gbf,
        float* __restrict__ invr)
{
    extern __shared__ char smem[];
    __bf16* sA = (__bf16*)smem;          // 16 x 1024 bf16 (32 KB), dead after k-loop
    float*  sC = (float*)smem;           // 16 x 3072 f32 (192 KB), live after barrier

    const int tid   = threadIdx.x;
    const int wave  = tid >> 5;
    const int lane  = tid & 31;
    const int half  = lane >> 4;
    const int mrow  = lane & 15;
    const size_t tokBase = (size_t)blockIdx.x * TOK;

    // cooperative A-stripe load: 16*1024 bf16 = 2048 x v8bf
    {
        const v8bf* src = (const v8bf*)(Eb + tokBase * D_);
        v8bf*       dst = (v8bf*)sA;
        for (int i = tid; i < 2048; i += THREADS) dst[i] = src[i];
    }
    __syncthreads();

    v8f acc[TILES_PER_WAVE];
    const v8f vzero = {0.f, 0.f, 0.f, 0.f, 0.f, 0.f, 0.f, 0.f};
#pragma unroll
    for (int j = 0; j < TILES_PER_WAVE; ++j) acc[j] = vzero;

    const int ntBase = wave * TILES_PER_WAVE;   // first N-tile of this wave

#pragma unroll 1
    for (int kt = 0; kt < KTILES; ++kt) {
        // A fragment straight from LDS per documented 16-bit A 16x32 layout
        const __bf16* arow = sA + mrow * D_ + kt * 32;
        v8bf a1 = *(const v8bf*)(arow + half * 8);
        v8bf a2 = *(const v8bf*)(arow + 16 + half * 8);
        v16bf a = __builtin_shufflevector(a1, a2, 0,1,2,3,4,5,6,7,
                                                  8,9,10,11,12,13,14,15);
        const __bf16* wbase = Wb + ((size_t)kt * NTILES + ntBase) * 512 + lane * 16;
        if (kt + 1 < KTILES) {
            // stream hint for next K-step's weight tiles (global_prefetch_b8)
            __builtin_prefetch(Wb + ((size_t)(kt + 1) * NTILES + ntBase) * 512, 0, 1);
        }
        // phase 1: issue ALL B-fragment loads (16 b128s in flight)
        v8bf b1[TILES_PER_WAVE], b2[TILES_PER_WAVE];
#pragma unroll
        for (int j = 0; j < TILES_PER_WAVE; ++j) {
            const __bf16* bp = wbase + (size_t)j * 512;
            b1[j] = *(const v8bf*)(bp);
            b2[j] = *(const v8bf*)(bp + 8);
        }
        // hard scheduling fence: nothing may cross (keeps all 16 loads issued
        // before the first WMMA -> backend emits partial s_wait_loadcnt)
        __builtin_amdgcn_sched_barrier(0);
        // phase 2: WMMA burst
#pragma unroll
        for (int j = 0; j < TILES_PER_WAVE; ++j) {
            v16bf b = __builtin_shufflevector(b1[j], b2[j], 0,1,2,3,4,5,6,7,
                                                            8,9,10,11,12,13,14,15);
            acc[j] = __builtin_amdgcn_wmma_f32_16x16x32_bf16(
                false, a, false, b, (short)0, acc[j], false, false);
        }
    }
    __syncthreads();   // everyone done reading sA before sC overwrites it

    // spill accumulators to LDS: C layout VGPR r -> M=r (lanes 0-15) / M=8+r
#pragma unroll
    for (int j = 0; j < TILES_PER_WAVE; ++j) {
        int col = (ntBase + j) * 16 + mrow;
#pragma unroll
        for (int r = 0; r < 8; ++r)
            sC[(half * 8 + r) * NTOT + col] = acc[j][r];
    }
    __syncthreads();

    // ---------------- fused epilogue: one wave per token -------------------
    if (wave < TOK) {
        const int t = wave;
        const size_t tok = tokBase + t;
        const float* hrow = hidden + tok * GD;
        const float* crow = sC + (size_t)t * NTOT;

        float ssK0 = 0.f, ssK1 = 0.f, ssQ0 = 0.f, ssQ1 = 0.f;
        float dot0 = 0.f, dot1 = 0.f, sv2 = 0.f;
        for (int d = lane; d < D_; d += 32) {
            float k0v = crow[d]        + key_bias[d];
            float k1v = crow[D_ + d]   + key_bias[D_ + d];
            float h0  = hrow[d];
            float h1  = hrow[D_ + d];
            float vv  = crow[GD + d]   + value_bias[d];
            ssK0 += k0v * k0v;  ssK1 += k1v * k1v;
            ssQ0 += h0 * h0;    ssQ1 += h1 * h1;
            dot0 += h0 * k0v * q_scale[d]       * k_scale[d];
            dot1 += h1 * k1v * q_scale[D_ + d]  * k_scale[D_ + d];
            sv2  += vv * vv;
        }
#pragma unroll
        for (int m = 16; m >= 1; m >>= 1) {
            ssK0 += __shfl_xor(ssK0, m, 32);
            ssK1 += __shfl_xor(ssK1, m, 32);
            ssQ0 += __shfl_xor(ssQ0, m, 32);
            ssQ1 += __shfl_xor(ssQ1, m, 32);
            dot0 += __shfl_xor(dot0, m, 32);
            dot1 += __shfl_xor(dot1, m, 32);
            sv2  += __shfl_xor(sv2,  m, 32);
        }
        const float inv_d = 1.f / 1024.f;
        float rk0 = rsqrtf(ssK0 * inv_d + 1e-6f);
        float rk1 = rsqrtf(ssK1 * inv_d + 1e-6f);
        float rq0 = rsqrtf(ssQ0 * inv_d + 1e-6f);
        float rq1 = rsqrtf(ssQ1 * inv_d + 1e-6f);
        float qk0 = dot0 * rk0 * rq0 * (1.f / 32.f);   // /sqrt(D)
        float qk1 = dot1 * rk1 * rq1 * (1.f / 32.f);
        float sg0 = (qk0 > 0.f) ? 1.f : ((qk0 < 0.f) ? -1.f : 0.f);
        float sg1 = (qk1 > 0.f) ? 1.f : ((qk1 < 0.f) ? -1.f : 0.f);
        float pre0 = sg0 * sqrtf(fmaxf(fabsf(qk0), 1e-6f));
        float pre1 = sg1 * sqrtf(fmaxf(fabsf(qk1), 1e-6f));
        float gate0 = 1.f / (1.f + expf(-pre0));
        float gate1 = 1.f / (1.f + expf(-pre1));
        // mean(gated^2) = gate^2 * mean(value^2) -> conv RMSNorm needs 1 scalar
        if (lane == 0) {
            invr[tok * 2 + 0] = rsqrtf(gate0 * gate0 * sv2 * inv_d + 1e-6f);
            invr[tok * 2 + 1] = rsqrtf(gate1 * gate1 * sv2 * inv_d + 1e-6f);
        }
        for (int d = lane; d < D_; d += 32) {
            float vv = crow[GD + d] + value_bias[d];
            float g0 = gate0 * vv;
            float g1 = gate1 * vv;
            out[tok * GD + d]       = g0;
            out[tok * GD + D_ + d]  = g1;
            gbf[tok * GD + d]       = (__bf16)g0;
            gbf[tok * GD + D_ + d]  = (__bf16)g1;
        }
    }
}

// ---------------------------------------------------------------------------
// K3: depthwise causal dilated conv (k=4, dil=3) + SiLU + residual.
// xn is rebuilt on the fly from the bf16 gated copy + per-(tok,g) inv_rms.
// One thread = 4 consecutive channels of one (b,s).
// ---------------------------------------------------------------------------
__global__ void k3_conv(const __bf16* __restrict__ gbf,
                        const float* __restrict__ invr,
                        const float* __restrict__ cns,
                        const float* __restrict__ ck,
                        float* __restrict__ out)
{
    int idx = blockIdx.x * blockDim.x + threadIdx.x;
    if (idx >= NTOK * (GD / 4)) return;
    int c4 = (idx & 511) * 4;            // channel base, multiple of 4
    int bs = idx >> 9;                   // token index b*4096+s
    int s  = bs & 4095;
    int g  = c4 >> 10;

    float y0 = 0.f, y1 = 0.f, y2 = 0.f, y3 = 0.f;
#pragma unroll
    for (int j = 0; j < 4; ++j) {
        int sp = s - 9 + 3 * j;          // causal taps: s-9, s-6, s-3, s
        if (sp >= 0) {
            int tk = bs - (s - sp);
            float ir = invr[(size_t)tk * 2 + g];
            const __bf16* xp = gbf + (size_t)tk * GD + c4;
            const float*  wp = ck + j * GD + c4;
            y0 += wp[0] * ((float)xp[0] * ir * cns[c4 + 0]);
            y1 += wp[1] * ((float)xp[1] * ir * cns[c4 + 1]);
            y2 += wp[2] * ((float)xp[2] * ir * cns[c4 + 2]);
            y3 += wp[3] * ((float)xp[3] * ir * cns[c4 + 3]);
        }
    }
    float4* op = (float4*)(out + (size_t)bs * GD + c4);
    float4 o = *op;
    o.x += y0 * (1.f / (1.f + expf(-y0)));
    o.y += y1 * (1.f / (1.f + expf(-y1)));
    o.z += y2 * (1.f / (1.f + expf(-y2)));
    o.w += y3 * (1.f / (1.f + expf(-y3)));
    *op = o;
}

// ---------------------------------------------------------------------------
// launch: workspace layout (bytes):
//   [0, 6.29M)       Wb   bf16 packed weights (1024x3072)
//   [6.29M, 39.8M)   Eb   bf16 gathered embeddings (16384x1024)
//   [39.8M, 39.98M)  invr f32 per-(token,g) conv inv-rms (16384x2)
//   [39.98M, 107M)   gbf  bf16 gated copy (16384x2048)
// ---------------------------------------------------------------------------
extern "C" void kernel_launch(void* const* d_in, const int* in_sizes, int n_in,
                              void* d_out, int out_size, void* d_ws, size_t ws_size,
                              hipStream_t stream)
{
    const float* hidden     = (const float*)d_in[0];
    const int*   ids        = (const int*)  d_in[1];
    const float* emb_table  = (const float*)d_in[2];
    const float* key_kernel = (const float*)d_in[3];
    const float* key_bias   = (const float*)d_in[4];
    const float* val_kernel = (const float*)d_in[5];
    const float* val_bias   = (const float*)d_in[6];
    const float* k_scale    = (const float*)d_in[7];
    const float* q_scale    = (const float*)d_in[8];
    const float* cns        = (const float*)d_in[9];
    const float* conv_k     = (const float*)d_in[10];
    float* out = (float*)d_out;

    char* ws = (char*)d_ws;
    __bf16* Wb   = (__bf16*)(ws);
    __bf16* Eb   = (__bf16*)(ws + 6291456);
    float*  invr = (float*) (ws + 39845888);
    __bf16* gbf  = (__bf16*)(ws + 39976960);

    // K0: weight pack (3.1M elems)
    k0_pack<<<(D_ * NTOT + 255) / 256, 256, 0, stream>>>(key_kernel, val_kernel, Wb);
    // K1: hashed embedding gather (one block per token)
    k1_gather<<<NTOK, 256, 0, stream>>>(ids, emb_table, Eb);
    // K2: WMMA GEMM + fused gating epilogue (192 KB dynamic LDS per block)
    k2_gemm<<<NTOK / TOK, THREADS, NTOT * TOK * sizeof(float), stream>>>(
        Eb, Wb, hidden, key_bias, val_bias, k_scale, q_scale, out, gbf, invr);
    // K3: dilated depthwise conv + SiLU + residual
    k3_conv<<<(NTOK * (GD / 4) + 255) / 256, 256, 0, stream>>>(
        gbf, invr, cns, conv_k, out);
}
